// GatedFusionModel_7584912245022
// MI455X (gfx1250) — matrix operand order repaired and checked by reading, not compiled, not run on hardware
//
#include <hip/hip_runtime.h>

#define NN 20000
#define EE 320000
#define PP 100000
#define ETOT (EE + NN)

typedef __attribute__((ext_vector_type(16))) __bf16 v16bf;
typedef __attribute__((ext_vector_type(8)))  float  v8f;
typedef __attribute__((ext_vector_type(4)))  unsigned v4u;
typedef __attribute__((ext_vector_type(8)))  int    v8i_t;
typedef __attribute__((ext_vector_type(4)))  int    v4i_t;

#if defined(__has_builtin)
#if __has_builtin(__builtin_amdgcn_tensor_load_to_lds) && \
    __has_builtin(__builtin_amdgcn_s_wait_tensorcnt)
#define HAVE_TDM 1
#endif
#endif
#ifndef HAVE_TDM
#define HAVE_TDM 0
#endif

union ABfrag { v16bf v; unsigned u[8]; uint4 q[2]; };

__device__ __forceinline__ unsigned short f2bf(float f) {
  unsigned u = __float_as_uint(f);
  u += 0x7FFFu + ((u >> 16) & 1u);
  return (unsigned short)(u >> 16);
}
// order-preserving float<->uint encoding for atomicMax-based segment max
__device__ __forceinline__ unsigned fenc(float f) {
  unsigned u = __float_as_uint(f);
  return (u & 0x80000000u) ? ~u : (u | 0x80000000u);
}
__device__ __forceinline__ float fdec(unsigned u) {
  unsigned b = (u & 0x80000000u) ? (u & 0x7FFFFFFFu) : ~u;
  return __uint_as_float(b);
}

#if HAVE_TDM
// Issue one 2-D TDM transfer: tile (tile_rows x tile_k bf16) from a row-major
// [*, K] bf16 matrix into LDS at lds_off.  D# layout per CDNA5 ISA ch.8:
//  group0: count=1 | lds_addr[63:32] | global_addr[120:64] | type=2[127:126]
//  group1: data_size=1(2B)[17:16] | tensor_dim0=K[79:48] | tensor_dim1=rows
//          [111:80] | tile_dim0[127:112] | tile_dim1[143:128] |
//          tensor_dim0_stride=K[207:160]
__device__ __forceinline__ void tdm_load_2d(const unsigned short* gptr,
                                            unsigned lds_off, int K, int rows,
                                            int tile_rows, int tile_k) {
  unsigned long long ga = (unsigned long long)(uintptr_t)gptr;
  unsigned uK = (unsigned)K;
  unsigned ur = (unsigned)(rows < 0 ? 0 : (rows > 65535 ? 65535 : rows));
  v4u g0 = {1u, lds_off, (unsigned)(ga & 0xFFFFFFFFu),
            (unsigned)((ga >> 32) & 0x01FFFFFFu) | 0x80000000u};
  v8i_t g1 = {(int)0x10000u,
              (int)(uK << 16),
              (int)((uK >> 16) | (ur << 16)),
              (int)((ur >> 16) | ((unsigned)tile_k << 16)),
              (int)(unsigned)tile_rows,
              (int)uK,
              0, 0};
  v4i_t z4 = {0, 0, 0, 0};
#if __clang_major__ >= 23
  v8i_t z8 = {0, 0, 0, 0, 0, 0, 0, 0};
  __builtin_amdgcn_tensor_load_to_lds(g0, g1, z4, z4, z8, 0);
#else
  __builtin_amdgcn_tensor_load_to_lds(g0, g1, z4, z4, 0);
#endif
}
#endif

// ---------------------------------------------------------------- utilities
__global__ void cvt_bf16_kernel(const float* __restrict__ in,
                                unsigned short* __restrict__ out, int n) {
  int i = blockIdx.x * blockDim.x + threadIdx.x;
  if (i < n) out[i] = f2bf(in[i]);
}
__global__ void fill_f32_kernel(float* __restrict__ p, float v, int n) {
  int i = blockIdx.x * blockDim.x + threadIdx.x;
  if (i < n) p[i] = v;
}
__global__ void fill_u32_kernel(unsigned* __restrict__ p, unsigned v, int n) {
  int i = blockIdx.x * blockDim.x + threadIdx.x;
  if (i < n) p[i] = v;
}
__global__ void fill_from_scalar_kernel(float* __restrict__ p,
                                        const float* __restrict__ v, int n) {
  int i = blockIdx.x * blockDim.x + threadIdx.x;
  if (i < n) p[i] = v[0];
}

// ------------------------------------------------- generic WMMA bf16 GEMM
// C[M,Nc] = A[M,K] @ B[K,Nc], bf16 inputs, f32 out.  K in {256,512}, Nc%64==0.
// Block = 128 thr = 4 waves; WG tile 128x64; wave tile 32x64 (2 A frags reuse
// each B frag).  A panel is streamed into LDS in 128x128 chunks by the Tensor
// Data Mover (double buffered, s_wait_tensorcnt pipelined); B column panel is
// staged transposed (BT[n][k]) so B fragments are contiguous dword pairs.
__global__ __launch_bounds__(128) void gemm_bf16_wmma(
    const unsigned short* __restrict__ A, const unsigned short* __restrict__ B,
    float* __restrict__ C, int M, int K, int Nc) {
  extern __shared__ unsigned short smem[];
  const int Kc = 128;
  unsigned short* AS = smem;                 // [2][128][Kc]
  unsigned short* BT = smem + 2 * 128 * Kc;  // [64][K]
  const int tid = threadIdx.x;
  const int cbase = blockIdx.y * 64;
  const int rowblock = blockIdx.x * 128;

  for (int idx = tid; idx < 64 * K; idx += 128) {
    int n = idx & 63;
    int k = idx >> 6;
    BT[n * K + k] = B[(size_t)k * Nc + cbase + n];
  }

  const int wave = tid >> 5;
  const int lane = tid & 31;
  const int t = lane & 15;
  const int hi = lane >> 4;
  const int nchunks = K / Kc;
  const int r0 = wave * 32 + t;        // local A rows for this lane's frags
  const int r1 = r0 + 16;

  v8f acc[2][4] = {};

#if HAVE_TDM
  if (wave == 0) {
    tdm_load_2d(A + (size_t)rowblock * K, (unsigned)(uintptr_t)AS, K,
                M - rowblock, 128, Kc);
  }
  for (int jc = 0; jc < nchunks; ++jc) {
    if (wave == 0) {
      if (jc + 1 < nchunks) {
        tdm_load_2d(A + (size_t)rowblock * K + (jc + 1) * Kc,
                    (unsigned)(uintptr_t)(AS + ((jc + 1) & 1) * 128 * Kc), K,
                    M - rowblock, 128, Kc);
        __builtin_amdgcn_s_wait_tensorcnt(1);
      } else {
        __builtin_amdgcn_s_wait_tensorcnt(0);
      }
    }
    __syncthreads();  // chunk jc resident in LDS for all waves
    const unsigned short* asb = AS + (jc & 1) * 128 * Kc;
#else
  for (int jc = 0; jc < nchunks; ++jc) {
    __syncthreads();  // previous chunk fully consumed before overwrite
    for (int idx = tid; idx < 128 * Kc; idx += 128) {
      int r = idx >> 7;
      int kl = idx & (Kc - 1);
      int gr = rowblock + r;
      if (gr >= M) gr = M - 1;
      AS[idx] = A[(size_t)gr * K + jc * Kc + kl];
    }
    __syncthreads();
    const unsigned short* asb = AS;
#endif
    for (int kl = 0; kl < Kc; kl += 32) {
      ABfrag a0, a1;
      {
        const unsigned short* p0 = asb + r0 * Kc + kl + hi * 8;
        const unsigned short* p1 = asb + r1 * Kc + kl + hi * 8;
        a0.q[0] = *(const uint4*)(p0);
        a0.q[1] = *(const uint4*)(p0 + 16);
        a1.q[0] = *(const uint4*)(p1);
        a1.q[1] = *(const uint4*)(p1 + 16);
      }
      int kabs = jc * Kc + kl;
#pragma unroll
      for (int nt = 0; nt < 4; ++nt) {
        ABfrag b;
        const unsigned short* Bcol = &BT[(nt * 16 + t) * K + kabs + hi * 8];
#pragma unroll
        for (int v = 0; v < 8; ++v) b.u[v] = *(const unsigned*)(Bcol + 2 * v + (v >= 4 ? 8 : 0));
        acc[0][nt] = __builtin_amdgcn_wmma_f32_16x16x32_bf16(
            false, a0.v, false, b.v, (short)0, acc[0][nt], false, false);
        acc[1][nt] = __builtin_amdgcn_wmma_f32_16x16x32_bf16(
            false, a1.v, false, b.v, (short)0, acc[1][nt], false, false);
      }
    }
#if HAVE_TDM
    __syncthreads();  // done reading this buffer before TDM refills it
#endif
  }

#pragma unroll
  for (int f = 0; f < 2; ++f)
#pragma unroll
    for (int nt = 0; nt < 4; ++nt) {
      int n = cbase + nt * 16 + t;
#pragma unroll
      for (int i = 0; i < 8; ++i) {
        int mm = rowblock + wave * 32 + f * 16 + hi * 8 + i;
        if (mm < M) C[(size_t)mm * Nc + n] = acc[f][nt][i];
      }
    }
}

// ------------------------------------ fused pair-gather predictor GEMM
// rows: pair[r] = [hf[u[r]] | hf[v[r]]] (bf16, 512); B = pw1b [512,512];
// epilogue: out[r] += relu(acc + pb1[n]) * pw2[n]   (out pre-init to pb2)
__global__ __launch_bounds__(128) void gemm_pred_wmma(
    const unsigned short* __restrict__ HF, const int* __restrict__ U,
    const int* __restrict__ V, const unsigned short* __restrict__ B,
    const float* __restrict__ pb1, const float* __restrict__ pw2,
    float* __restrict__ out, int P) {
  __shared__ unsigned short BT[64 * 512];
  const int tid = threadIdx.x;
  const int cbase = blockIdx.y * 64;
  const int K = 512, Nc = 512;
  for (int idx = tid; idx < 64 * K; idx += 128) {
    int n = idx & 63;
    int k = idx >> 6;
    BT[n * K + k] = B[(size_t)k * Nc + cbase + n];
  }
  __syncthreads();

  const int wave = tid >> 5;
  const int lane = tid & 31;
  const int t = lane & 15;
  const int hi = lane >> 4;
  const int rbase = blockIdx.x * 128 + wave * 32;
  int r0 = rbase + t;
  int r1 = rbase + 16 + t;
  int r0c = r0 < P ? r0 : P - 1;
  int r1c = r1 < P ? r1 : P - 1;
  const unsigned short* uRow0 = HF + (size_t)U[r0c] * 256;
  const unsigned short* vRow0 = HF + (size_t)V[r0c] * 256;
  const unsigned short* uRow1 = HF + (size_t)U[r1c] * 256;
  const unsigned short* vRow1 = HF + (size_t)V[r1c] * 256;

  v8f acc[2][4] = {};
  for (int k = 0; k < K; k += 32) {
    const unsigned short* A0 = (k < 256) ? (uRow0 + k) : (vRow0 + (k - 256));
    const unsigned short* A1 = (k < 256) ? (uRow1 + k) : (vRow1 + (k - 256));
    ABfrag a0, a1;
    a0.q[0] = *(const uint4*)(A0 + hi * 8);
    a0.q[1] = *(const uint4*)(A0 + hi * 8 + 16);
    a1.q[0] = *(const uint4*)(A1 + hi * 8);
    a1.q[1] = *(const uint4*)(A1 + hi * 8 + 16);
#pragma unroll
    for (int nt = 0; nt < 4; ++nt) {
      ABfrag b;
      const unsigned short* Bcol = &BT[(nt * 16 + t) * K + k + hi * 8];
#pragma unroll
      for (int v = 0; v < 8; ++v) b.u[v] = *(const unsigned*)(Bcol + 2 * v + (v >= 4 ? 8 : 0));
      acc[0][nt] = __builtin_amdgcn_wmma_f32_16x16x32_bf16(
          false, a0.v, false, b.v, (short)0, acc[0][nt], false, false);
      acc[1][nt] = __builtin_amdgcn_wmma_f32_16x16x32_bf16(
          false, a1.v, false, b.v, (short)0, acc[1][nt], false, false);
    }
  }
#pragma unroll
  for (int f = 0; f < 2; ++f)
#pragma unroll
    for (int nt = 0; nt < 4; ++nt) {
      int n = cbase + nt * 16 + t;
      float w2 = pw2[n];
      float bb = pb1[n];
#pragma unroll
      for (int i = 0; i < 8; ++i) {
        int rr = rbase + f * 16 + hi * 8 + i;
        if (rr < P) {
          float val = acc[f][nt][i] + bb;
          val = fmaxf(val, 0.0f);
          atomicAdd(&out[rr], val * w2);
        }
      }
    }
}

// ------------------------------------------------------ GAT edge kernels
__global__ void attn_coef_kernel(const float* __restrict__ h,
                                 const float* __restrict__ a_s,
                                 const float* __restrict__ a_d,
                                 float* __restrict__ outs,
                                 float* __restrict__ outd, int Nn, int Hh) {
  int idx = blockIdx.x * blockDim.x + threadIdx.x;
  if (idx >= Nn * Hh) return;
  int i = idx / Hh;
  int hh = idx - i * Hh;
  int C = 256 / Hh;
  const float* hr = h + (size_t)i * 256 + hh * C;
  const float* asr = a_s + hh * C;
  const float* adr = a_d + hh * C;
  float s1 = 0.f, s2 = 0.f;
  for (int c = 0; c < C; ++c) {
    float hv = hr[c];
    s1 += hv * asr[c];
    s2 += hv * adr[c];
  }
  outs[idx] = s1;
  outd[idx] = s2;
}

__global__ void edge_logits_kernel(const int* __restrict__ ei,
                                   const float* __restrict__ as_,
                                   const float* __restrict__ ad_,
                                   float* __restrict__ ee,
                                   unsigned* __restrict__ emax, int Hh) {
  int idx = blockIdx.x * blockDim.x + threadIdx.x;
  if (idx >= ETOT * Hh) return;
  int e = idx / Hh;
  int hh = idx - e * Hh;
  int s = (e < EE) ? ei[e] : (e - EE);
  int d = (e < EE) ? ei[EE + e] : (e - EE);
  float xv = as_[s * Hh + hh] + ad_[d * Hh + hh];
  float lr = xv > 0.f ? xv : 0.2f * xv;
  ee[(size_t)e * Hh + hh] = lr;
  atomicMax(&emax[d * Hh + hh], fenc(lr));
}

__global__ void edge_norm_kernel(const int* __restrict__ ei,
                                 float* __restrict__ ee,
                                 const unsigned* __restrict__ emax,
                                 float* __restrict__ den, int Hh) {
  int idx = blockIdx.x * blockDim.x + threadIdx.x;
  if (idx >= ETOT * Hh) return;
  int e = idx / Hh;
  int hh = idx - e * Hh;
  int d = (e < EE) ? ei[EE + e] : (e - EE);
  float p = __expf(ee[idx] - fdec(emax[d * Hh + hh]));
  ee[idx] = p;
  atomicAdd(&den[d * Hh + hh], p);
}

__global__ __launch_bounds__(256) void edge_aggregate_kernel(
    const int* __restrict__ ei, const float* __restrict__ ee,
    const float* __restrict__ den, const float* __restrict__ feat,
    float* __restrict__ agg, int Hh) {
  int e = blockIdx.x;
  int c = threadIdx.x;  // 0..255
  int s = (e < EE) ? ei[e] : (e - EE);
  int d = (e < EE) ? ei[EE + e] : (e - EE);
  int hh = (c * Hh) >> 8;
  float alpha = ee[(size_t)e * Hh + hh] / fmaxf(den[d * Hh + hh], 1e-16f);
  atomicAdd(&agg[(size_t)d * 256 + c], feat[(size_t)s * 256 + c] * alpha);
}

__global__ void bias_elu_kernel(const float* __restrict__ agg,
                                const float* __restrict__ b1,
                                unsigned short* __restrict__ h1b, int n) {
  int idx = blockIdx.x * blockDim.x + threadIdx.x;
  if (idx >= n) return;
  int c = idx & 255;
  float v = agg[idx] + b1[c];
  float h = v > 0.f ? v : (__expf(v) - 1.0f);
  h1b[idx] = f2bf(h);
}

__global__ void bias2_gatein_kernel(float* __restrict__ hg,
                                    const float* __restrict__ b2,
                                    const float* __restrict__ x,
                                    unsigned short* __restrict__ gateb, int n) {
  int idx = blockIdx.x * blockDim.x + threadIdx.x;
  if (idx >= n) return;
  int i = idx >> 8;
  int c = idx & 255;
  float v = hg[idx] + b2[c];
  hg[idx] = v;
  gateb[(size_t)i * 512 + c] = f2bf(x[idx]);
  gateb[(size_t)i * 512 + 256 + c] = f2bf(v);
}

__global__ void gate_kernel(const float* __restrict__ t1,
                            const float* __restrict__ gb1,
                            const float* __restrict__ gw2,
                            const float* __restrict__ gb2,
                            float* __restrict__ g, int Nn) {
  int i = blockIdx.x * blockDim.x + threadIdx.x;
  if (i >= Nn) return;
  float s = gb2[0];
  const float* tr = t1 + (size_t)i * 256;
  for (int j = 0; j < 256; ++j) {
    float v = tr[j] + gb1[j];
    v = fmaxf(v, 0.f);
    s += v * gw2[j];
  }
  g[i] = 1.0f / (1.0f + __expf(-s));
}

__global__ void hfinal_kernel(const float* __restrict__ x,
                              const float* __restrict__ hg,
                              const float* __restrict__ g,
                              unsigned short* __restrict__ hfb, int n) {
  int idx = blockIdx.x * blockDim.x + threadIdx.x;
  if (idx >= n) return;
  int i = idx >> 8;
  float gg = g[i];
  float hf = (1.0f - gg) * x[idx] + gg * hg[idx];
  hfb[idx] = f2bf(hf);
}

// ---------------------------------------------------------------- launcher
extern "C" void kernel_launch(void* const* d_in, const int* in_sizes, int n_in,
                              void* d_out, int out_size, void* d_ws,
                              size_t ws_size, hipStream_t stream) {
  const float* x   = (const float*)d_in[0];
  const int*   ei  = (const int*)d_in[1];
  const int*   un  = (const int*)d_in[2];
  const int*   vn  = (const int*)d_in[3];
  const float* W1  = (const float*)d_in[4];
  const float* a1s = (const float*)d_in[5];
  const float* a1d = (const float*)d_in[6];
  const float* b1  = (const float*)d_in[7];
  const float* W2  = (const float*)d_in[8];
  const float* a2s = (const float*)d_in[9];
  const float* a2d = (const float*)d_in[10];
  const float* b2  = (const float*)d_in[11];
  const float* gw1 = (const float*)d_in[12];
  const float* gb1 = (const float*)d_in[13];
  const float* gw2 = (const float*)d_in[14];
  const float* gb2 = (const float*)d_in[15];
  const float* pw1 = (const float*)d_in[16];
  const float* pb1 = (const float*)d_in[17];
  const float* pw2 = (const float*)d_in[18];
  const float* pb2 = (const float*)d_in[19];
  float* out = (float*)d_out;

  char* ws = (char*)d_ws;
  size_t off = 0;
  auto carve = [&](size_t bytes) {
    char* p = ws + off;
    off += (bytes + 255) & ~(size_t)255;
    return p;
  };
  unsigned short* xb    = (unsigned short*)carve((size_t)NN * 256 * 2);
  unsigned short* w1b   = (unsigned short*)carve(256 * 256 * 2);
  unsigned short* w2b   = (unsigned short*)carve(256 * 256 * 2);
  unsigned short* gw1b  = (unsigned short*)carve(512 * 256 * 2);
  unsigned short* pw1b  = (unsigned short*)carve(512 * 512 * 2);
  float*          hpre  = (float*)carve((size_t)NN * 256 * 4);  // later: t1
  float*          agg1  = (float*)carve((size_t)NN * 256 * 4);
  unsigned short* h1b   = (unsigned short*)carve((size_t)NN * 256 * 2);  // later: hfb
  float*          h2pre = (float*)carve((size_t)NN * 256 * 4);
  float*          hg    = (float*)carve((size_t)NN * 256 * 4);  // agg2 -> h_graph
  unsigned short* gateb = (unsigned short*)carve((size_t)NN * 512 * 2);
  float*          as1   = (float*)carve((size_t)NN * 4 * 4);
  float*          ad1   = (float*)carve((size_t)NN * 4 * 4);
  unsigned*       emax1 = (unsigned*)carve((size_t)NN * 4 * 4);
  float*          den1  = (float*)carve((size_t)NN * 4 * 4);
  float*          ee1   = (float*)carve((size_t)ETOT * 4 * 4);
  float*          as2   = (float*)carve((size_t)NN * 4);
  float*          ad2   = (float*)carve((size_t)NN * 4);
  unsigned*       emax2 = (unsigned*)carve((size_t)NN * 4);
  float*          den2  = (float*)carve((size_t)NN * 4);
  float*          ee2   = (float*)carve((size_t)ETOT * 4);
  float*          gbuf  = (float*)carve((size_t)NN * 4);

  const unsigned NEG_INF_ENC = 0x007FFFFFu;  // fenc(-inf)
  auto blks = [](int n, int b) { return (n + b - 1) / b; };
  auto shmem = [](int K) { return (size_t)(2 * 128 * 128 + 64 * K) * 2; };

  // ---- precision conversion of GEMM operands
  cvt_bf16_kernel<<<blks(NN * 256, 256), 256, 0, stream>>>(x, xb, NN * 256);
  cvt_bf16_kernel<<<blks(256 * 256, 256), 256, 0, stream>>>(W1, w1b, 256 * 256);
  cvt_bf16_kernel<<<blks(256 * 256, 256), 256, 0, stream>>>(W2, w2b, 256 * 256);
  cvt_bf16_kernel<<<blks(512 * 256, 256), 256, 0, stream>>>(gw1, gw1b, 512 * 256);
  cvt_bf16_kernel<<<blks(512 * 512, 256), 256, 0, stream>>>(pw1, pw1b, 512 * 512);

  // ---- init accumulators
  fill_f32_kernel<<<blks(NN * 256, 256), 256, 0, stream>>>(agg1, 0.f, NN * 256);
  fill_f32_kernel<<<blks(NN * 256, 256), 256, 0, stream>>>(hg, 0.f, NN * 256);
  fill_f32_kernel<<<blks(NN * 4, 256), 256, 0, stream>>>(den1, 0.f, NN * 4);
  fill_f32_kernel<<<blks(NN, 256), 256, 0, stream>>>(den2, 0.f, NN);
  fill_u32_kernel<<<blks(NN * 4, 256), 256, 0, stream>>>(emax1, NEG_INF_ENC, NN * 4);
  fill_u32_kernel<<<blks(NN, 256), 256, 0, stream>>>(emax2, NEG_INF_ENC, NN);

  // ---- GAT layer 1
  gemm_bf16_wmma<<<dim3(blks(NN, 128), 4), 128, shmem(256), stream>>>(
      xb, w1b, hpre, NN, 256, 256);
  attn_coef_kernel<<<blks(NN * 4, 256), 256, 0, stream>>>(hpre, a1s, a1d, as1, ad1, NN, 4);
  edge_logits_kernel<<<blks(ETOT * 4, 256), 256, 0, stream>>>(ei, as1, ad1, ee1, emax1, 4);
  edge_norm_kernel<<<blks(ETOT * 4, 256), 256, 0, stream>>>(ei, ee1, emax1, den1, 4);
  edge_aggregate_kernel<<<ETOT, 256, 0, stream>>>(ei, ee1, den1, hpre, agg1, 4);
  bias_elu_kernel<<<blks(NN * 256, 256), 256, 0, stream>>>(agg1, b1, h1b, NN * 256);

  // ---- GAT layer 2 (1 head, 256 channels)
  gemm_bf16_wmma<<<dim3(blks(NN, 128), 4), 128, shmem(256), stream>>>(
      h1b, w2b, h2pre, NN, 256, 256);
  attn_coef_kernel<<<blks(NN, 256), 256, 0, stream>>>(h2pre, a2s, a2d, as2, ad2, NN, 1);
  edge_logits_kernel<<<blks(ETOT, 256), 256, 0, stream>>>(ei, as2, ad2, ee2, emax2, 1);
  edge_norm_kernel<<<blks(ETOT, 256), 256, 0, stream>>>(ei, ee2, emax2, den2, 1);
  edge_aggregate_kernel<<<ETOT, 256, 0, stream>>>(ei, ee2, den2, h2pre, hg, 1);
  bias2_gatein_kernel<<<blks(NN * 256, 256), 256, 0, stream>>>(hg, b2, x, gateb, NN * 256);

  // ---- gate MLP + fusion
  float* t1 = hpre;  // reuse
  gemm_bf16_wmma<<<dim3(blks(NN, 128), 4), 128, shmem(512), stream>>>(
      gateb, gw1b, t1, NN, 512, 256);
  gate_kernel<<<blks(NN, 128), 128, 0, stream>>>(t1, gb1, gw2, gb2, gbuf, NN);
  unsigned short* hfb = h1b;  // reuse
  hfinal_kernel<<<blks(NN * 256, 256), 256, 0, stream>>>(x, hg, gbuf, hfb, NN * 256);

  // ---- pair predictor (gather-fused GEMM + relu-GEMV epilogue into d_out)
  fill_from_scalar_kernel<<<blks(PP, 256), 256, 0, stream>>>(out, pb2, PP);
  gemm_pred_wmma<<<dim3(blks(PP, 128), 8), 128, 0, stream>>>(hfb, un, vn, pw1b, pb1, pw2, out, PP);
}
